// PatchReprogrammingEncoder_2370821947796
// MI455X (gfx1250) — compile-verified, matrix operand order checked by reading, not run on hardware
//
#include <hip/hip_runtime.h>
#include <hip/hip_bf16.h>
#include <stdint.h>

// ---------------- problem constants ----------------
#define NTOK   15
#define NTOKP  16      // padded rows per token (row 15 zeroed)
#define DP     128     // D_PATCH
#define DM     768     // D_PLM
#define NHEAD  4
#define HD     192     // head dim
#define NPROTO 64
#define EPS    1e-5f
#define INV_SQRT_HD 0.07216878364870323f  // 1/sqrt(192)

typedef __bf16 bf16_t;
typedef __attribute__((ext_vector_type(16))) __bf16 v16bf;
typedef __attribute__((ext_vector_type(8)))  float  v8f;
typedef __attribute__((ext_vector_type(4)))  int    v4i;

#define DEV static __device__ __forceinline__

// round-to-nearest-even float -> bf16
DEV bf16_t f2bf(float x) {
  union { float f; uint32_t u; } v; v.f = x;
  uint32_t r = (v.u + 0x7FFFu + ((v.u >> 16) & 1u)) >> 16;
  union { uint16_t u; bf16_t b; } o; o.u = (uint16_t)r;
  return o.b;
}

// ------------- async global->LDS staging (CDNA5 ASYNCcnt path) --------------
#if defined(__gfx1250__) && __has_builtin(__builtin_amdgcn_global_load_async_to_lds_b128)
#define HAVE_ASYNC_LDS 1
#else
#define HAVE_ASYNC_LDS 0
#endif

DEV void async_copy_b128(const void* g, void* l) {
#if HAVE_ASYNC_LDS
  __builtin_amdgcn_global_load_async_to_lds_b128(
      (__attribute__((address_space(1))) v4i*)g,
      (__attribute__((address_space(3))) v4i*)l, 0, 0);
#else
  *(uint4*)l = *(const uint4*)g;
#endif
}
DEV void async_copy_wait() {
#if HAVE_ASYNC_LDS
#if __has_builtin(__builtin_amdgcn_s_wait_asynccnt)
  __builtin_amdgcn_s_wait_asynccnt(0);
#else
  asm volatile("s_wait_asynccnt 0x0" ::: "memory");
#endif
#endif
}

// ---------------- WMMA fragment loaders (self-consistent A/B mapping) -------
// A: 16x32 bf16, row-major with leading dim lda.
// lane l: row = m0 + (l&15), 16 contiguous k values starting at k0 + (l>>4)*16
DEV v16bf load_a(const bf16_t* A, int lda, int lane, int m0, int k0) {
  int r  = m0 + (lane & 15);
  int kb = k0 + ((lane >> 4) << 4);
  return *(const v16bf*)(A + r * lda + kb);
}
// B: 32x16 bf16 taken from a K-major (i.e. B^T row-major) buffer with leading dim ldk.
// lane l: col = n0 + (l&15), 16 contiguous k values starting at k0 + (l>>4)*16
DEV v16bf load_bT(const bf16_t* BT, int ldk, int lane, int n0, int k0) {
  int c  = n0 + (lane & 15);
  int kb = k0 + ((lane >> 4) << 4);
  return *(const v16bf*)(BT + c * ldk + kb);
}
DEV v8f wmma_bf16(v16bf a, v16bf b, v8f c) {
  return __builtin_amdgcn_wmma_f32_16x16x32_bf16(false, a, false, b, (short)0, c,
                                                 false, false);
}

// ---------------- kernel 1: fp32 -> bf16 transposed weights -----------------
__global__ void cvt_T_kernel(const float* __restrict__ src, bf16_t* __restrict__ dst,
                             int R, int C) {
  int idx = blockIdx.x * blockDim.x + threadIdx.x;
  if (idx >= R * C) return;
  int r = idx / C, c = idx - r * C;
  dst[c * R + r] = f2bf(src[idx]);     // dst is [C][R] = src^T
}

// ---------------- kernel 2: prototype K/V -----------------------------------
// Kmat [p][768] bf16 (k-dim contiguous per proto col)
// VmatT [d][64] bf16 (k-dim = p contiguous per output col d)
__global__ void kv_kernel(const float* __restrict__ protos,
                          const float* __restrict__ wk, const float* __restrict__ bk,
                          const float* __restrict__ wv, const float* __restrict__ bv,
                          bf16_t* __restrict__ Kmat, bf16_t* __restrict__ VmatT) {
  int idx = blockIdx.x * blockDim.x + threadIdx.x;
  if (idx >= NPROTO * DM) return;
  int p = idx / DM, d = idx - p * DM;
  float ak = bk[d], av = bv[d];
  const float* pr = protos + p * DM;
  for (int e = 0; e < DM; ++e) {
    float pe = pr[e];
    ak = fmaf(pe, wk[e * DM + d], ak);
    av = fmaf(pe, wv[e * DM + d], av);
  }
  Kmat[p * DM + d]      = f2bf(ak);
  VmatT[d * NPROTO + p] = f2bf(av);
}

// ---------------- kernel 3: patch embeddings + LayerNorm --------------------
DEV void reduce2_128(float& a, float& b, float* red) {
  #pragma unroll
  for (int m = 16; m >= 1; m >>= 1) {
    a += __shfl_xor(a, m, 32);
    b += __shfl_xor(b, m, 32);
  }
  int w = threadIdx.x >> 5;
  if ((threadIdx.x & 31) == 0) { red[w * 2] = a; red[w * 2 + 1] = b; }
  __syncthreads();
  a = red[0] + red[2] + red[4] + red[6];
  b = red[1] + red[3] + red[5] + red[7];
  __syncthreads();
}

__global__ void pe_kernel(const float* __restrict__ state,
                          const float* __restrict__ w_scalar, const float* __restrict__ b_scalar,
                          const float* __restrict__ w_patch,  const float* __restrict__ b_patch,
                          const float* __restrict__ feat_emb,
                          const float* __restrict__ pn_g, const float* __restrict__ pn_b,
                          bf16_t* __restrict__ PE) {
  __shared__ float ss[36];
  __shared__ float red[8];
  const int d = threadIdx.x;       // 0..127
  const int n = blockIdx.x;
  if (d < 36) ss[d] = state[n * 36 + d];
  __syncthreads();

  float pv[NTOK];
  // scalar rows: features 0,1 -> rows 0,1 ; feature 5 -> row 14
  pv[0]  = ss[0 * 6 + 5] * w_scalar[0 * DP + d] + b_scalar[0 * DP + d] + feat_emb[0 * DP + d];
  pv[1]  = ss[1 * 6 + 5] * w_scalar[1 * DP + d] + b_scalar[1 * DP + d] + feat_emb[1 * DP + d];
  pv[14] = ss[5 * 6 + 5] * w_scalar[2 * DP + d] + b_scalar[2 * DP + d] + feat_emb[5 * DP + d];
  // patch rows: features 2,3,4 -> rows 2..13 (4 windows each, window width 3)
  #pragma unroll
  for (int f = 0; f < 3; ++f) {
    #pragma unroll
    for (int k = 0; k < 4; ++k) {
      float a = 0.f;
      #pragma unroll
      for (int p = 0; p < 3; ++p)
        a = fmaf(ss[(2 + f) * 6 + k + p], w_patch[(f * 3 + p) * DP + d], a);
      pv[2 + f * 4 + k] = a + b_patch[f * DP + d] + feat_emb[(2 + f) * DP + d];
    }
  }
  // per-row LayerNorm over d=128 and bf16 store
  bf16_t* dst = PE + (size_t)n * NTOKP * DP;
  for (int r = 0; r < NTOK; ++r) {
    float v = pv[r];
    float s = v, s2 = v * v;
    reduce2_128(s, s2, red);
    float mean = s * (1.f / DP);
    float var  = s2 * (1.f / DP) - mean * mean;
    float nv   = (v - mean) * rsqrtf(var + EPS) * pn_g[d] + pn_b[d];
    dst[r * DP + d] = f2bf(nv);
  }
  dst[15 * DP + d] = f2bf(0.f);    // zero pad row
}

// ---------------- kernel 4: Q projection + attention + rep ------------------
// one workgroup (8 wave32) per token
__global__ void attn_kernel(const bf16_t* __restrict__ PE,
                            const bf16_t* __restrict__ wqT, const float* __restrict__ bq,
                            const bf16_t* __restrict__ Kmat, const bf16_t* __restrict__ VmatT,
                            bf16_t* __restrict__ REP) {
  __shared__ alignas(32) bf16_t pe_s[NTOKP * DP];        //  4 KB
  __shared__ alignas(32) bf16_t q_s[NTOKP * DM];         // 24 KB
  __shared__ alignas(32) float  sc_s[NHEAD * NTOKP * NPROTO]; // 16 KB
  __shared__ alignas(32) bf16_t at_s[NHEAD * NTOKP * NPROTO]; //  8 KB

  const int n    = blockIdx.x;
  const int tid  = threadIdx.x;
  const int wave = tid >> 5;
  const int lane = tid & 31;
  const int mh   = (lane >> 4) << 3;   // 0 or 8: row-half of C fragments
  const int nc   = lane & 15;          // C fragment column within tile

  // stage PE tile (16x128 bf16 = 4 KB) -> 256 x 16B, async direct-to-LDS
  async_copy_b128(((const uint4*)(PE + (size_t)n * NTOKP * DP)) + tid,
                  ((uint4*)pe_s) + tid);
  async_copy_wait();
  __syncthreads();

  // ---- Q = PE @ wq + bq : M=16, N=768 (96 cols/wave), K=128 ----
  {
    v8f acc[6]; v8f z = {};
    #pragma unroll
    for (int i = 0; i < 6; ++i) acc[i] = z;
    #pragma unroll
    for (int kt = 0; kt < 4; ++kt) {
      v16bf a = load_a(pe_s, DP, lane, 0, kt * 32);
      #pragma unroll
      for (int i = 0; i < 6; ++i) {
        v16bf b = load_bT(wqT, DP, lane, wave * 96 + i * 16, kt * 32);
        acc[i] = wmma_bf16(a, b, acc[i]);
      }
    }
    #pragma unroll
    for (int i = 0; i < 6; ++i) {
      int col = wave * 96 + i * 16 + nc;
      float bias = bq[col];
      #pragma unroll
      for (int j = 0; j < 8; ++j)
        q_s[(mh + j) * DM + col] = f2bf(acc[i][j] + bias);
    }
  }
  __syncthreads();

  // ---- scores = Q . K^T / sqrt(hd) : 4 heads x (16x64), K=192 ----
  #pragma unroll
  for (int pass = 0; pass < 2; ++pass) {
    int tt = wave + pass * 8;          // 16 tiles
    int h  = tt >> 2, pt = tt & 3;
    v8f acc = {};
    #pragma unroll
    for (int kt = 0; kt < 6; ++kt) {
      v16bf a = load_a(q_s, DM, lane, 0, h * HD + kt * 32);
      v16bf b = load_bT(Kmat, DM, lane, pt * 16, h * HD + kt * 32);
      acc = wmma_bf16(a, b, acc);
    }
    #pragma unroll
    for (int j = 0; j < 8; ++j)
      sc_s[(h * NTOKP + mh + j) * NPROTO + pt * 16 + nc] = acc[j] * INV_SQRT_HD;
  }
  __syncthreads();

  // ---- softmax over 64 prototypes (60 live rows) ----
  if (tid < 60) {
    int h = tid / NTOK, t = tid - h * NTOK;
    const float* row = sc_s + (h * NTOKP + t) * NPROTO;
    float mx = row[0];
    for (int p = 1; p < NPROTO; ++p) mx = fmaxf(mx, row[p]);
    float s = 0.f;
    for (int p = 0; p < NPROTO; ++p) s += __expf(row[p] - mx);
    float inv = 1.f / s;
    bf16_t* arow = at_s + (h * NTOKP + t) * NPROTO;
    for (int p = 0; p < NPROTO; ++p) arow[p] = f2bf(__expf(row[p] - mx) * inv);
  } else if (tid < 64) {
    int h = tid - 60;                  // zero pad row 15 per head
    bf16_t* arow = at_s + (h * NTOKP + 15) * NPROTO;
    for (int p = 0; p < NPROTO; ++p) arow[p] = f2bf(0.f);
  }
  __syncthreads();

  // ---- rep = attn @ V : 4 heads x (16x192), K=64 -> 48 tiles, 6/wave ----
  bf16_t* rout = REP + (size_t)n * NTOKP * DM;
  #pragma unroll
  for (int ti = 0; ti < 6; ++ti) {
    int tt = wave * 6 + ti;
    int h = tt / 12, dt = tt - h * 12;
    v8f acc = {};
    #pragma unroll
    for (int kt = 0; kt < 2; ++kt) {
      v16bf a = load_a(at_s + h * NTOKP * NPROTO, NPROTO, lane, 0, kt * 32);
      v16bf b = load_bT(VmatT, NPROTO, lane, h * HD + dt * 16, kt * 32);
      acc = wmma_bf16(a, b, acc);
    }
    int col = h * HD + dt * 16 + nc;
    #pragma unroll
    for (int j = 0; j < 8; ++j)
      rout[(mh + j) * DM + col] = f2bf(acc[j]);  // pad row stays 0 (attn row 15 == 0)
  }
}

// ---------------- kernel 5: out = rep@wo + pe@wp + biases, LayerNorm --------
// 2 tokens per workgroup (M=32), 8 wave32; each wave owns 96 output columns
__global__ void out_kernel(const bf16_t* __restrict__ REP, const bf16_t* __restrict__ PE,
                           const bf16_t* __restrict__ woT, const bf16_t* __restrict__ wpT,
                           const float* __restrict__ bo, const float* __restrict__ bp,
                           const float* __restrict__ on_g, const float* __restrict__ on_b,
                           float* __restrict__ out) {
  __shared__ alignas(32) char smem[98304];        // 96 KB, reused
  bf16_t* rep_s = (bf16_t*)smem;                  // 32x768 bf16 = 48 KB
  bf16_t* pe_s  = (bf16_t*)(smem + 49152);        // 32x128 bf16 =  8 KB
  float*  ob    = (float*)smem;                   // 32x768 f32  = 96 KB (after GEMM)

  const int tid  = threadIdx.x;
  const int wave = tid >> 5;
  const int lane = tid & 31;
  const int mh   = (lane >> 4) << 3;
  const int nc   = lane & 15;
  const int tok0 = blockIdx.x * 2;

  // stage 2 tokens of REP (contiguous) and PE, async direct-to-LDS
  {
    const uint4* r4 = (const uint4*)(REP + (size_t)tok0 * NTOKP * DM);
    #pragma unroll
    for (int it = 0; it < 12; ++it)
      async_copy_b128(r4 + tid + it * 256, ((uint4*)rep_s) + tid + it * 256);
    const uint4* p4 = (const uint4*)(PE + (size_t)tok0 * NTOKP * DP);
    #pragma unroll
    for (int it = 0; it < 2; ++it)
      async_copy_b128(p4 + tid + it * 256, ((uint4*)pe_s) + tid + it * 256);
    async_copy_wait();
  }
  __syncthreads();

  v8f acc[2][6]; v8f z = {};
  #pragma unroll
  for (int mi = 0; mi < 2; ++mi)
    #pragma unroll
    for (int i = 0; i < 6; ++i) acc[mi][i] = z;

  // rep @ wo : K = 768, software-pipelined B fragments (1 K-step lookahead)
  {
    v16bf bc[6];
    #pragma unroll
    for (int i = 0; i < 6; ++i)
      bc[i] = load_bT(woT, DM, lane, wave * 96 + i * 16, 0);
    for (int kt = 0; kt < 24; ++kt) {
      v16bf bn[6];
      const bool more = (kt + 1) < 24;
      if (more) {
        #pragma unroll
        for (int i = 0; i < 6; ++i)
          bn[i] = load_bT(woT, DM, lane, wave * 96 + i * 16, (kt + 1) * 32);
      }
      v16bf a0 = load_a(rep_s, DM, lane, 0,  kt * 32);
      v16bf a1 = load_a(rep_s, DM, lane, 16, kt * 32);
      #pragma unroll
      for (int i = 0; i < 6; ++i) {
        acc[0][i] = wmma_bf16(a0, bc[i], acc[0][i]);
        acc[1][i] = wmma_bf16(a1, bc[i], acc[1][i]);
      }
      if (more) {
        #pragma unroll
        for (int i = 0; i < 6; ++i) bc[i] = bn[i];
      }
    }
  }
  // pe @ wp : K = 128
  #pragma unroll
  for (int kt = 0; kt < 4; ++kt) {
    v16bf a0 = load_a(pe_s, DP, lane, 0,  kt * 32);
    v16bf a1 = load_a(pe_s, DP, lane, 16, kt * 32);
    #pragma unroll
    for (int i = 0; i < 6; ++i) {
      v16bf b = load_bT(wpT, DP, lane, wave * 96 + i * 16, kt * 32);
      acc[0][i] = wmma_bf16(a0, b, acc[0][i]);
      acc[1][i] = wmma_bf16(a1, b, acc[1][i]);
    }
  }
  __syncthreads();   // done with rep_s/pe_s, smem becomes ob

  #pragma unroll
  for (int i = 0; i < 6; ++i) {
    int col = wave * 96 + i * 16 + nc;
    float bias = bo[col] + bp[col];
    #pragma unroll
    for (int mi = 0; mi < 2; ++mi)
      #pragma unroll
      for (int j = 0; j < 8; ++j)
        ob[(mi * 16 + mh + j) * DM + col] = acc[mi][i][j] + bias;
  }
  __syncthreads();

  // row-wise LayerNorm over 768, one wave per row, skip pad rows
  for (int rr = 0; rr < 4; ++rr) {
    int r = wave + rr * 8;
    int tok = tok0 + (r >> 4);
    int t = r & 15;
    if (t == 15) continue;
    const float* row = ob + r * DM;
    float s = 0.f, s2 = 0.f;
    for (int c = lane; c < DM; c += 32) { float x = row[c]; s += x; s2 += x * x; }
    #pragma unroll
    for (int m = 16; m >= 1; m >>= 1) {
      s  += __shfl_xor(s,  m, 32);
      s2 += __shfl_xor(s2, m, 32);
    }
    float mean = s * (1.f / DM);
    float var  = s2 * (1.f / DM) - mean * mean;
    float inv  = rsqrtf(var + EPS);
    float* orow = out + ((size_t)tok * NTOK + t) * DM;
    for (int c = lane; c < DM; c += 32)
      orow[c] = (row[c] - mean) * inv * on_g[c] + on_b[c];
  }
}

// ---------------- host launcher ---------------------------------------------
extern "C" void kernel_launch(void* const* d_in, const int* in_sizes, int n_in,
                              void* d_out, int out_size, void* d_ws, size_t ws_size,
                              hipStream_t stream) {
  // setup_inputs order:
  const float* state    = (const float*)d_in[0];
  const float* w_scalar = (const float*)d_in[1];
  const float* b_scalar = (const float*)d_in[2];
  const float* w_patch  = (const float*)d_in[3];
  const float* b_patch  = (const float*)d_in[4];
  const float* feat_emb = (const float*)d_in[5];
  const float* pn_g     = (const float*)d_in[6];
  const float* pn_b     = (const float*)d_in[7];
  const float* wq       = (const float*)d_in[8];
  const float* bq       = (const float*)d_in[9];
  const float* wk       = (const float*)d_in[10];
  const float* bk       = (const float*)d_in[11];
  const float* wv       = (const float*)d_in[12];
  const float* bv       = (const float*)d_in[13];
  const float* wp       = (const float*)d_in[14];
  const float* bp       = (const float*)d_in[15];
  const float* wo       = (const float*)d_in[16];
  const float* bo       = (const float*)d_in[17];
  const float* on_g     = (const float*)d_in[18];
  const float* on_b     = (const float*)d_in[19];
  const float* protos   = (const float*)d_in[20];
  float* out = (float*)d_out;

  const int N = in_sizes[0] / 36;   // B*T tokens
  if (N <= 0) return;

  // workspace layout (bf16), offsets 256B-aligned
  char* ws = (char*)d_ws;
  size_t off = 0;
  bf16_t* wqT  = (bf16_t*)(ws + off); off += (size_t)DM * DP * 2;        // [768][128]
  bf16_t* wpT  = (bf16_t*)(ws + off); off += (size_t)DM * DP * 2;        // [768][128]
  bf16_t* woT  = (bf16_t*)(ws + off); off += (size_t)DM * DM * 2;        // [768][768]
  bf16_t* Kmat = (bf16_t*)(ws + off); off += (size_t)NPROTO * DM * 2;    // [64][768]
  bf16_t* VmT  = (bf16_t*)(ws + off); off += (size_t)DM * NPROTO * 2;    // [768][64]
  bf16_t* PE   = (bf16_t*)(ws + off); off += (size_t)N * NTOKP * DP * 2; // [N][16][128]
  bf16_t* REP  = (bf16_t*)(ws + off); off += (size_t)N * NTOKP * DM * 2; // [N][16][768]

  cvt_T_kernel<<<(DP * DM + 255) / 256, 256, 0, stream>>>(wq, wqT, DP, DM);
  cvt_T_kernel<<<(DP * DM + 255) / 256, 256, 0, stream>>>(wp, wpT, DP, DM);
  cvt_T_kernel<<<(DM * DM + 255) / 256, 256, 0, stream>>>(wo, woT, DM, DM);
  kv_kernel<<<(NPROTO * DM + 255) / 256, 256, 0, stream>>>(protos, wk, bk, wv, bv, Kmat, VmT);
  pe_kernel<<<N, 128, 0, stream>>>(state, w_scalar, b_scalar, w_patch, b_patch,
                                   feat_emb, pn_g, pn_b, PE);
  attn_kernel<<<N, 256, 0, stream>>>(PE, wqT, bq, Kmat, VmT, REP);
  out_kernel<<<N / 2, 256, 0, stream>>>(REP, PE, woT, wpT, bo, bp, on_g, on_b, out);
}
